// LinearAttentionModel_80676665688088
// MI455X (gfx1250) — matrix operand order repaired and checked by reading, not compiled, not run on hardware
//
#include <hip/hip_runtime.h>
#include <hip/hip_bf16.h>
#include <math.h>

// ---------------------------------------------------------------------------
// RWKV-7 block for MI455X (gfx1250, wave32, WMMA).
// GEMMs (97% of FLOPs) -> v_wmma_f32_16x16x32_bf16, double-buffered LDS tiles
// filled with global_load_async_to_lds_b128 (ASYNCcnt), weights pre-transposed
// to bf16 so both A and B fragments are contiguous ds_load_b128 pairs.
// WKV7 scan -> 64-thread blocks, one state row per thread in registers.
// ---------------------------------------------------------------------------

#ifndef USE_ASYNC_COPY
#define USE_ASYNC_COPY 1
#endif

typedef __bf16 bf16t;
typedef __attribute__((ext_vector_type(16))) __bf16 v16bf;
typedef __attribute__((ext_vector_type(8)))  float  v8f;

#define B_SZ 2
#define T_SZ 2048
#define D_SZ 1024
#define H_SZ 16
#define N_SZ 64
#define DF_SZ 4096
#define M_SZ (B_SZ * T_SZ)   // 4096 rows

#define EPS_LN 1e-5f
#define EPS_GN 64e-5f

// ------------------------------- helpers -----------------------------------

__device__ __forceinline__ float sigmoidf_(float x) { return 1.f / (1.f + expf(-x)); }

__device__ __forceinline__ void copy16B(const void* g, void* l) {
#if USE_ASYNC_COPY
  unsigned lds = (unsigned)(size_t)l;                  // low 32 bits = LDS addr
  unsigned long long ga = (unsigned long long)(size_t)g;
  asm volatile("global_load_async_to_lds_b128 %0, %1, off"
               :: "v"(lds), "v"(ga) : "memory");
#else
  *(float4*)l = *(const float4*)g;
#endif
}

__device__ __forceinline__ void lds_wait_async() {
#if USE_ASYNC_COPY
  asm volatile("s_wait_asynccnt 0" ::: "memory");
#endif
}

// tree reduce over n threads (n = power of two, all threads participate)
__device__ __forceinline__ float block_reduce(float v, float* red, int tid, int n) {
  red[tid] = v;
  __syncthreads();
  for (int s = n >> 1; s > 0; s >>= 1) {
    if (tid < s) red[tid] += red[tid + s];
    __syncthreads();
  }
  float r = red[0];
  __syncthreads();
  return r;
}

// ---------------- fp32 [K,N] -> bf16 transpose [N,K] (weights) --------------

__global__ __launch_bounds__(256, 1)
void transpose_f32_bf16_kernel(const float* __restrict__ src, bf16t* __restrict__ dst,
                               int K, int N) {
  __shared__ float tile[32][33];
  const int bn = blockIdx.x * 32;   // N base
  const int bk = blockIdx.y * 32;   // K base
  const int tx = threadIdx.x & 31, ty = threadIdx.x >> 5;  // 32x8
  for (int i = ty; i < 32; i += 8) {
    const int k = bk + i, n = bn + tx;
    tile[i][tx] = (k < K && n < N) ? src[(size_t)k * N + n] : 0.f;
  }
  __syncthreads();
  for (int i = ty; i < 32; i += 8) {
    const int n = bn + i, k = bk + tx;
    if (n < N && k < K) dst[(size_t)n * K + k] = (bf16t)tile[tx][i];
  }
}

// ------------------------------- LayerNorm ----------------------------------

__global__ __launch_bounds__(256, 1)
void ln_kernel(const float* __restrict__ x, const float* __restrict__ w,
               const float* __restrict__ b, float* __restrict__ y) {
  __shared__ float red[256];
  const int row = blockIdx.x, tid = threadIdx.x;
  const float* xr = x + (size_t)row * D_SZ;
  float s = 0.f;
  for (int c = tid; c < D_SZ; c += 256) s += xr[c];
  const float mean = block_reduce(s, red, tid, 256) * (1.f / D_SZ);
  float s2 = 0.f;
  for (int c = tid; c < D_SZ; c += 256) { float d = xr[c] - mean; s2 += d * d; }
  const float var = block_reduce(s2, red, tid, 256) * (1.f / D_SZ);
  const float inv = rsqrtf(var + EPS_LN);
  float* yr = y + (size_t)row * D_SZ;
  for (int c = tid; c < D_SZ; c += 256) yr[c] = (xr[c] - mean) * inv * w[c] + b[c];
}

// ----------------------- token-shift mixing (6-way / 1-way) -----------------

__global__ __launch_bounds__(256, 1)
void mix6_kernel(const float* __restrict__ xn,
                 const float* cr, const float* cw, const float* ck,
                 const float* cv, const float* ca, const float* cg,
                 bf16t* xr, bf16t* xw, bf16t* xk, bf16t* xv, bf16t* xa, bf16t* xg) {
  const size_t total = (size_t)M_SZ * D_SZ;
  for (size_t idx = (size_t)blockIdx.x * 256 + threadIdx.x; idx < total;
       idx += (size_t)gridDim.x * 256) {
    const int row = (int)(idx / D_SZ), col = (int)(idx % D_SZ);
    const float cur  = xn[idx];
    const float prev = (row % T_SZ) ? xn[idx - D_SZ] : 0.f;
    const float dx = prev - cur;
    xr[idx] = (bf16t)(cur + dx * cr[col]);
    xw[idx] = (bf16t)(cur + dx * cw[col]);
    xk[idx] = (bf16t)(cur + dx * ck[col]);
    xv[idx] = (bf16t)(cur + dx * cv[col]);
    xa[idx] = (bf16t)(cur + dx * ca[col]);
    xg[idx] = (bf16t)(cur + dx * cg[col]);
  }
}

__global__ __launch_bounds__(256, 1)
void mix1_kernel(const float* __restrict__ xn, const float* c, bf16t* out) {
  const size_t total = (size_t)M_SZ * D_SZ;
  for (size_t idx = (size_t)blockIdx.x * 256 + threadIdx.x; idx < total;
       idx += (size_t)gridDim.x * 256) {
    const int row = (int)(idx / D_SZ), col = (int)(idx % D_SZ);
    const float cur  = xn[idx];
    const float prev = (row % T_SZ) ? xn[idx - D_SZ] : 0.f;
    out[idx] = (bf16t)(cur + (prev - cur) * c[col]);
  }
}

// --------------------------- WMMA bf16 GEMM ---------------------------------
// C[M,N] = epilogue(A[M,K] @ B[K,N]); A bf16 row-major [M,K], B given as
// transposed bf16 BT[N,K]. Block: 128 threads = 4 waves (2x2), block tile
// 128x128, wave tile 64x64 = 4x4 wmma 16x16x32 tiles. Double-buffered LDS,
// async global->LDS copies, one s_wait_asynccnt + one barrier per K-step.

enum {
  EPI_F32 = 0, EPI_BF16, EPI_TANH_BF16, EPI_SIGMOID_BF16,
  EPI_BIAS_SIGMOID_F32, EPI_DECAY_F32, EPI_RELUSQ_BF16, EPI_ADDRES_F32
};

template <int EPI>
__global__ __launch_bounds__(128, 1)
void gemm_bf16_wmma(const bf16t* __restrict__ A, const bf16t* __restrict__ BT,
                    void* __restrict__ Cout,
                    const float* __restrict__ bias,   // per output column
                    const float* __restrict__ Res,    // residual [M,N]
                    int Md, int Nd, int Kd) {
  __shared__ bf16t sA[2][128][32];
  __shared__ bf16t sB[2][128][32];
  const int tid  = threadIdx.x;
  const int lane = tid & 31;
  const int wave = tid >> 5;        // 0..3
  const int wm   = wave >> 1;       // 0..1  (M sub-block)
  const int wn   = wave & 1;        // 0..1  (N sub-block)
  const int half = lane >> 4;       // 0..1
  const int l16  = lane & 15;
  const int blockN = blockIdx.x * 128;
  const int blockM = blockIdx.y * 128;

  // Each of the 128 threads owns one row of sA and one row of sB.
  const bf16t* gA = A  + (size_t)(blockM + tid) * Kd;
  const bf16t* gB = BT + (size_t)(blockN + tid) * Kd;
  const bool bvalid = (blockN + tid) < Nd;

  auto load_tile = [&](int buf, int k0) {
    bf16t* dA = &sA[buf][tid][0];
    copy16B(gA + k0,      dA);
    copy16B(gA + k0 + 8,  dA + 8);
    copy16B(gA + k0 + 16, dA + 16);
    copy16B(gA + k0 + 24, dA + 24);
    bf16t* dB = &sB[buf][tid][0];
    if (bvalid) {
      copy16B(gB + k0,      dB);
      copy16B(gB + k0 + 8,  dB + 8);
      copy16B(gB + k0 + 16, dB + 16);
      copy16B(gB + k0 + 24, dB + 24);
    } else {                       // zero-fill out-of-range N rows (LoRA dims)
      const float4 z = {0.f, 0.f, 0.f, 0.f};
      float4* d4 = (float4*)dB;
      d4[0] = z; d4[1] = z; d4[2] = z; d4[3] = z;
    }
    if (k0 + 32 < Kd) {            // L2 prefetch of the K+2 slice
      __builtin_prefetch(gA + k0 + 32, 0, 1);
      __builtin_prefetch(gB + k0 + 32, 0, 1);
    }
  };

  const v8f vzero = {0.f, 0.f, 0.f, 0.f, 0.f, 0.f, 0.f, 0.f};
  v8f acc[4][4];
#pragma unroll
  for (int mt = 0; mt < 4; ++mt)
#pragma unroll
    for (int nt = 0; nt < 4; ++nt) acc[mt][nt] = vzero;

  load_tile(0, 0);
  const int nk = Kd >> 5;
  for (int it = 0; it < nk; ++it) {
    const int buf = it & 1;
    lds_wait_async();     // only outstanding asyncs are this tile's
    __syncthreads();      // all waves' tile data visible
    if (it + 1 < nk) load_tile(buf ^ 1, (it + 1) << 5);

    // Fragments per ISA 7.12.2/7.12.4 16-bit layouts (all contiguous runs):
    //   A lane: K in [8*half,+8) and [16+8*half,+8)   -> 2x b128
    //   B lane: K in [8*half,+8) and [16+8*half,+8)   -> 2x b128
    union FragU { v16bf v; uint4 q[2]; };
    FragU af[4], bfr[4];
#pragma unroll
    for (int mt = 0; mt < 4; ++mt) {
      const int mrow = wm * 64 + mt * 16 + l16;
      af[mt].q[0] = *(const uint4*)&sA[buf][mrow][8 * half];
      af[mt].q[1] = *(const uint4*)&sA[buf][mrow][16 + 8 * half];
    }
#pragma unroll
    for (int nt = 0; nt < 4; ++nt) {
      const int nrow = wn * 64 + nt * 16 + l16;
      bfr[nt].q[0] = *(const uint4*)&sB[buf][nrow][8 * half];
      bfr[nt].q[1] = *(const uint4*)&sB[buf][nrow][16 + 8 * half];
    }
#pragma unroll
    for (int mt = 0; mt < 4; ++mt)
#pragma unroll
      for (int nt = 0; nt < 4; ++nt)
        acc[mt][nt] = __builtin_amdgcn_wmma_f32_16x16x32_bf16(
            false, af[mt].v, false, bfr[nt].v, (short)0, acc[mt][nt], false, false);
  }

  // epilogue: C layout -> vgpr e: row = base + 8*half + e, col = base + l16
#pragma unroll
  for (int mt = 0; mt < 4; ++mt) {
#pragma unroll
    for (int nt = 0; nt < 4; ++nt) {
      const int col = blockN + wn * 64 + nt * 16 + l16;
      if (col >= Nd) continue;
#pragma unroll
      for (int e = 0; e < 8; ++e) {
        const int row = blockM + wm * 64 + mt * 16 + half * 8 + e;
        if (row >= Md) continue;
        const size_t off = (size_t)row * Nd + col;
        const float v = acc[mt][nt][e];
        if (EPI == EPI_F32) {
          ((float*)Cout)[off] = v;
        } else if (EPI == EPI_BF16) {
          ((bf16t*)Cout)[off] = (bf16t)v;
        } else if (EPI == EPI_TANH_BF16) {
          ((bf16t*)Cout)[off] = (bf16t)tanhf(v);
        } else if (EPI == EPI_SIGMOID_BF16) {
          ((bf16t*)Cout)[off] = (bf16t)sigmoidf_(v);
        } else if (EPI == EPI_BIAS_SIGMOID_F32) {
          ((float*)Cout)[off] = sigmoidf_(bias[col] + v);
        } else if (EPI == EPI_DECAY_F32) {
          // w_log = -softplus(-(w0+z)) - 0.5 ; w = exp(-exp(w_log))
          const float y  = bias[col] + v;
          const float wl = -log1pf(expf(-y)) - 0.5f;
          ((float*)Cout)[off] = expf(-expf(wl));
        } else if (EPI == EPI_RELUSQ_BF16) {
          const float r = fmaxf(v, 0.f);
          ((bf16t*)Cout)[off] = (bf16t)(r * r);
        } else if (EPI == EPI_ADDRES_F32) {
          ((float*)Cout)[off] = Res[off] + v;
        }
      }
    }
  }
}

// ---------------- k / kk / a post-processing (per head L2 norm) -------------

__global__ __launch_bounds__(64, 1)
void post_kv_kernel(float* __restrict__ k, const float* __restrict__ a,
                    const float* __restrict__ kkc, const float* __restrict__ kac,
                    float* __restrict__ aa, float* __restrict__ bb) {
  __shared__ float red[64];
  const int row = blockIdx.x, h = blockIdx.y, i = threadIdx.x;
  const int c = h * N_SZ + i;
  const size_t idx = (size_t)row * D_SZ + c;
  const float kv = k[idx], av = a[idx];
  const float kk = kv * kkc[c];
  const float ss = block_reduce(kk * kk, red, i, 64);
  const float inv = 1.f / fmaxf(sqrtf(ss), 1e-12f);
  const float kkn = kk * inv;
  aa[idx] = -kkn;                         // a-input of WKV7
  bb[idx] = kkn * av;                     // b-input of WKV7
  k[idx]  = kv * (1.f + (av - 1.f) * kac[c]);
}

// --------------------------- WKV7 delta-rule scan ---------------------------
// One block per (b,h); thread i owns state row S[i][0..63] in registers.

__global__ __launch_bounds__(64, 1)
void wkv7_kernel(const float* __restrict__ r, const float* __restrict__ w,
                 const float* __restrict__ k, const float* __restrict__ v,
                 const float* __restrict__ aa, const float* __restrict__ bb,
                 float* __restrict__ o) {
  const int bh = blockIdx.x;
  const int b = bh / H_SZ, h = bh % H_SZ;
  const int i = threadIdx.x;
  __shared__ float sr[64], sw[64], sk[64], sv[64], sa[64], sb[64];
  float S[64];
#pragma unroll
  for (int j = 0; j < 64; ++j) S[j] = 0.f;
  const size_t base = ((size_t)b * T_SZ) * D_SZ + h * N_SZ + i;
  for (int t = 0; t < T_SZ; ++t) {
    const size_t idx = base + (size_t)t * D_SZ;
    sr[i] = r[idx]; sw[i] = w[idx]; sk[i] = k[idx];
    sv[i] = v[idx]; sa[i] = aa[idx]; sb[i] = bb[idx];
    __syncthreads();
    float sdota = 0.f;
#pragma unroll
    for (int j = 0; j < 64; ++j) sdota += S[j] * sa[j];
    const float vi = sv[i];
    float out = 0.f;
#pragma unroll
    for (int j = 0; j < 64; ++j) {
      const float s = S[j] * sw[j] + sdota * sb[j] + vi * sk[j];
      S[j] = s;
      out += s * sr[j];
    }
    o[idx] = out;
    __syncthreads();
  }
}

// ------------------ per-head GroupNorm + bonus + gate -----------------------

__global__ __launch_bounds__(64, 1)
void gn_bonus_gate_kernel(const float* __restrict__ o, const float* __restrict__ r,
                          const float* __restrict__ k, const float* __restrict__ v,
                          const float* __restrict__ g, const float* __restrict__ rk,
                          const float* __restrict__ lnxw, const float* __restrict__ lnxb,
                          bf16t* __restrict__ og) {
  __shared__ float red[64];
  const int row = blockIdx.x, h = blockIdx.y, i = threadIdx.x;
  const int c = h * N_SZ + i;
  const size_t idx = (size_t)row * D_SZ + c;
  const float ov = o[idx];
  const float mean = block_reduce(ov, red, i, 64) * (1.f / 64.f);
  const float d0 = ov - mean;
  const float var = block_reduce(d0 * d0, red, i, 64) * (1.f / 64.f);
  float y = d0 * rsqrtf(var + EPS_GN) * lnxw[c] + lnxb[c];
  const float bonus = block_reduce(r[idx] * k[idx] * rk[c], red, i, 64);
  y += bonus * v[idx];
  og[idx] = (bf16t)(y * g[idx]);
}

// ------------------------------- launcher -----------------------------------

extern "C" void kernel_launch(void* const* d_in, const int* in_sizes, int n_in,
                              void* d_out, int out_size, void* d_ws, size_t ws_size,
                              hipStream_t stream) {
  (void)in_sizes; (void)n_in; (void)out_size; (void)ws_size;

  // Inputs: setup_inputs() dict order (hidden_states, then params dict order)
  const float* hs     = (const float*)d_in[0];   // [B,T,D]
  const float* ln1_w  = (const float*)d_in[1];
  const float* ln1_b  = (const float*)d_in[2];
  const float* ln2_w  = (const float*)d_in[3];
  const float* ln2_b  = (const float*)d_in[4];
  const float* x_r    = (const float*)d_in[5];
  const float* x_w    = (const float*)d_in[6];
  const float* x_k    = (const float*)d_in[7];
  const float* x_v    = (const float*)d_in[8];
  const float* x_a    = (const float*)d_in[9];
  const float* x_g    = (const float*)d_in[10];
  const float* w0     = (const float*)d_in[11];
  const float* w1     = (const float*)d_in[12];  // [D,64]
  const float* w2     = (const float*)d_in[13];  // [64,D]
  const float* a0     = (const float*)d_in[14];
  const float* a1     = (const float*)d_in[15];  // [D,64]
  const float* a2     = (const float*)d_in[16];  // [64,D]
  const float* g1     = (const float*)d_in[17];  // [D,128]
  const float* g2     = (const float*)d_in[18];  // [128,D]
  const float* k_k    = (const float*)d_in[19];
  const float* k_a    = (const float*)d_in[20];
  const float* r_k    = (const float*)d_in[21];  // [H,N]
  const float* Wr     = (const float*)d_in[22];
  const float* Wk     = (const float*)d_in[23];
  const float* Wv     = (const float*)d_in[24];
  const float* Wo     = (const float*)d_in[25];
  const float* lnx_w  = (const float*)d_in[26];
  const float* lnx_b  = (const float*)d_in[27];
  const float* xk_ffn = (const float*)d_in[28];
  const float* Wk_ffn = (const float*)d_in[29];  // [D,4D]
  const float* Wv_ffn = (const float*)d_in[30];  // [4D,D]
  float* out = (float*)d_out;

  const size_t MD = (size_t)M_SZ * D_SZ;
  size_t off = 0;
  auto alloc = [&](size_t bytes) -> void* {
    void* p = (char*)d_ws + off;
    off += (bytes + 255) & ~(size_t)255;
    return p;
  };

  // bf16 transposed weight copies (BT[N,K])
  bf16t* tWr  = (bf16t*)alloc((size_t)D_SZ * D_SZ * 2);
  bf16t* tWk  = (bf16t*)alloc((size_t)D_SZ * D_SZ * 2);
  bf16t* tWv  = (bf16t*)alloc((size_t)D_SZ * D_SZ * 2);
  bf16t* tWo  = (bf16t*)alloc((size_t)D_SZ * D_SZ * 2);
  bf16t* tWkf = (bf16t*)alloc((size_t)D_SZ * DF_SZ * 2);
  bf16t* tWvf = (bf16t*)alloc((size_t)DF_SZ * D_SZ * 2);
  bf16t* tw1  = (bf16t*)alloc((size_t)D_SZ * 64 * 2);
  bf16t* tw2  = (bf16t*)alloc((size_t)64 * D_SZ * 2);
  bf16t* ta1  = (bf16t*)alloc((size_t)D_SZ * 64 * 2);
  bf16t* ta2  = (bf16t*)alloc((size_t)64 * D_SZ * 2);
  bf16t* tg1  = (bf16t*)alloc((size_t)D_SZ * 128 * 2);
  bf16t* tg2  = (bf16t*)alloc((size_t)128 * D_SZ * 2);
  // activations
  float* xn   = (float*)alloc(MD * 4);           // LN output (reused for ln2)
  bf16t* mxr  = (bf16t*)alloc(MD * 2);
  bf16t* mxw  = (bf16t*)alloc(MD * 2);
  bf16t* mxk  = (bf16t*)alloc(MD * 2);
  bf16t* mxv  = (bf16t*)alloc(MD * 2);
  bf16t* mxa  = (bf16t*)alloc(MD * 2);
  bf16t* mxg  = (bf16t*)alloc(MD * 2);
  float* rbuf = (float*)alloc(MD * 4);
  float* kbuf = (float*)alloc(MD * 4);
  float* vbuf = (float*)alloc(MD * 4);
  bf16t* tw_  = (bf16t*)alloc((size_t)M_SZ * 64 * 2);   // tanh(xw@w1)
  bf16t* ta_  = (bf16t*)alloc((size_t)M_SZ * 64 * 2);   // xa@a1
  bf16t* zg   = (bf16t*)alloc((size_t)M_SZ * 128 * 2);  // sigmoid(xg@g1)
  float* wdec = (float*)alloc(MD * 4);
  float* abuf = (float*)alloc(MD * 4);
  float* gbuf = (float*)alloc(MD * 4);
  float* aabf = (float*)alloc(MD * 4);
  float* bbbf = (float*)alloc(MD * 4);
  float* obuf = (float*)alloc(MD * 4);
  bf16t* og   = (bf16t*)alloc(MD * 2);
  float* xmid = (float*)alloc(MD * 4);
  bf16t* kf   = (bf16t*)alloc(MD * 2);
  bf16t* hffn = (bf16t*)alloc((size_t)M_SZ * DF_SZ * 2);

  auto cvtT = [&](const float* s, bf16t* d, int K, int N) {
    dim3 grid((N + 31) / 32, (K + 31) / 32);
    transpose_f32_bf16_kernel<<<grid, 256, 0, stream>>>(s, d, K, N);
  };
  cvtT(Wr, tWr, D_SZ, D_SZ);      cvtT(Wk, tWk, D_SZ, D_SZ);
  cvtT(Wv, tWv, D_SZ, D_SZ);      cvtT(Wo, tWo, D_SZ, D_SZ);
  cvtT(Wk_ffn, tWkf, D_SZ, DF_SZ);
  cvtT(Wv_ffn, tWvf, DF_SZ, D_SZ);
  cvtT(w1, tw1, D_SZ, 64);        cvtT(w2, tw2, 64, D_SZ);
  cvtT(a1, ta1, D_SZ, 64);        cvtT(a2, ta2, 64, D_SZ);
  cvtT(g1, tg1, D_SZ, 128);       cvtT(g2, tg2, 128, D_SZ);

  auto gemm = [&](int epi, const bf16t* A, const bf16t* BT, void* C,
                  const float* bias, const float* Res, int Md, int Nd, int Kd) {
    dim3 grid((Nd + 127) / 128, (Md + 127) / 128);
    switch (epi) {
      case EPI_F32:             gemm_bf16_wmma<EPI_F32><<<grid, 128, 0, stream>>>(A, BT, C, bias, Res, Md, Nd, Kd); break;
      case EPI_BF16:            gemm_bf16_wmma<EPI_BF16><<<grid, 128, 0, stream>>>(A, BT, C, bias, Res, Md, Nd, Kd); break;
      case EPI_TANH_BF16:       gemm_bf16_wmma<EPI_TANH_BF16><<<grid, 128, 0, stream>>>(A, BT, C, bias, Res, Md, Nd, Kd); break;
      case EPI_SIGMOID_BF16:    gemm_bf16_wmma<EPI_SIGMOID_BF16><<<grid, 128, 0, stream>>>(A, BT, C, bias, Res, Md, Nd, Kd); break;
      case EPI_BIAS_SIGMOID_F32:gemm_bf16_wmma<EPI_BIAS_SIGMOID_F32><<<grid, 128, 0, stream>>>(A, BT, C, bias, Res, Md, Nd, Kd); break;
      case EPI_DECAY_F32:       gemm_bf16_wmma<EPI_DECAY_F32><<<grid, 128, 0, stream>>>(A, BT, C, bias, Res, Md, Nd, Kd); break;
      case EPI_RELUSQ_BF16:     gemm_bf16_wmma<EPI_RELUSQ_BF16><<<grid, 128, 0, stream>>>(A, BT, C, bias, Res, Md, Nd, Kd); break;
      default:                  gemm_bf16_wmma<EPI_ADDRES_F32><<<grid, 128, 0, stream>>>(A, BT, C, bias, Res, Md, Nd, Kd); break;
    }
  };

  // ---- time mixing ----
  ln_kernel<<<M_SZ, 256, 0, stream>>>(hs, ln1_w, ln1_b, xn);
  mix6_kernel<<<4096, 256, 0, stream>>>(xn, x_r, x_w, x_k, x_v, x_a, x_g,
                                        mxr, mxw, mxk, mxv, mxa, mxg);

  gemm(EPI_F32, mxr, tWr, rbuf, nullptr, nullptr, M_SZ, D_SZ, D_SZ);
  gemm(EPI_F32, mxk, tWk, kbuf, nullptr, nullptr, M_SZ, D_SZ, D_SZ);
  gemm(EPI_F32, mxv, tWv, vbuf, nullptr, nullptr, M_SZ, D_SZ, D_SZ);

  gemm(EPI_TANH_BF16, mxw, tw1, tw_, nullptr, nullptr, M_SZ, 64, D_SZ);
  gemm(EPI_DECAY_F32, tw_, tw2, wdec, w0, nullptr, M_SZ, D_SZ, 64);

  gemm(EPI_BF16, mxa, ta1, ta_, nullptr, nullptr, M_SZ, 64, D_SZ);
  gemm(EPI_BIAS_SIGMOID_F32, ta_, ta2, abuf, a0, nullptr, M_SZ, D_SZ, 64);

  gemm(EPI_SIGMOID_BF16, mxg, tg1, zg, nullptr, nullptr, M_SZ, 128, D_SZ);
  gemm(EPI_F32, zg, tg2, gbuf, nullptr, nullptr, M_SZ, D_SZ, 128);

  post_kv_kernel<<<dim3(M_SZ, H_SZ), 64, 0, stream>>>(kbuf, abuf, k_k, k_a, aabf, bbbf);

  wkv7_kernel<<<B_SZ * H_SZ, 64, 0, stream>>>(rbuf, wdec, kbuf, vbuf, aabf, bbbf, obuf);

  gn_bonus_gate_kernel<<<dim3(M_SZ, H_SZ), 64, 0, stream>>>(
      obuf, rbuf, kbuf, vbuf, gbuf, r_k, lnx_w, lnx_b, og);

  gemm(EPI_ADDRES_F32, og, tWo, xmid, nullptr, hs, M_SZ, D_SZ, D_SZ);

  // ---- channel mixing ----
  ln_kernel<<<M_SZ, 256, 0, stream>>>(xmid, ln2_w, ln2_b, xn);
  mix1_kernel<<<4096, 256, 0, stream>>>(xn, xk_ffn, kf);
  gemm(EPI_RELUSQ_BF16, kf, tWkf, hffn, nullptr, nullptr, M_SZ, DF_SZ, D_SZ);
  gemm(EPI_ADDRES_F32, hffn, tWvf, out, nullptr, xmid, M_SZ, D_SZ, DF_SZ);
}
